// FastAttention_80745385165379
// MI455X (gfx1250) — matrix operand order repaired and checked by reading, not compiled, not run on hardware
//
#include <hip/hip_runtime.h>

typedef __attribute__((ext_vector_type(16))) __bf16 v16bf;
typedef __attribute__((ext_vector_type(8)))  float  v8f;
typedef unsigned int v4u __attribute__((ext_vector_type(4)));
typedef int          v8i __attribute__((ext_vector_type(8)));
typedef int          v4i __attribute__((ext_vector_type(4)));

#define DMODEL 1024
#define NHEADS 16
#define DK     64
#define BATCH  4
#define SEQ    2048
#define MTOT   (BATCH * SEQ) /* 8192 */

#if defined(__HIP_DEVICE_COMPILE__) && __has_builtin(__builtin_amdgcn_tensor_load_to_lds)
#define HAS_TDM 1
#else
#define HAS_TDM 0
#endif

__device__ __forceinline__ unsigned short f2bf(float f) {
  unsigned int u = __builtin_bit_cast(unsigned int, f);
  u += 0x7FFFu + ((u >> 16) & 1u);
  return (unsigned short)(u >> 16);
}

union Frag16 { v16bf bf; unsigned int u[8]; };

__device__ __forceinline__ v8f wmma_bf16(const Frag16& a, const Frag16& b, v8f c) {
  return __builtin_amdgcn_wmma_f32_16x16x32_bf16(false, a.bf, false, b.bf,
                                                 (short)0, c, false, false);
}

// ---- CDNA5 async global->LDS (ASYNCcnt) ----
__device__ __forceinline__ void async_b128(unsigned lds_off, const void* g) {
  asm volatile("global_load_async_to_lds_b128 %0, %1, off"
               :: "v"(lds_off), "v"((unsigned long long)g) : "memory");
}
__device__ __forceinline__ void wait_async0() {
  asm volatile("s_wait_asynccnt 0x0" ::: "memory");
}
__device__ __forceinline__ void wait_ds0() {
  asm volatile("s_wait_dscnt 0x0" ::: "memory");
}

#if HAS_TDM
// ---- Tensor Data Mover: 2D bf16 tile Global->LDS (D# per ISA 8.3/8.4) ----
__device__ __forceinline__ void tdm_load_2d(unsigned lds_off, const void* gaddr,
                                            unsigned tensor_d0, unsigned tensor_d1,
                                            unsigned tile_d0, unsigned tile_d1,
                                            unsigned long long stride0) {
  unsigned long long ga = (unsigned long long)gaddr;
  v4u g0;
  g0[0] = 1u;                                            // count=1 (valid), user mode
  g0[1] = lds_off;                                       // lds_addr (bytes)
  g0[2] = (unsigned)ga;                                  // global_addr[31:0]
  g0[3] = (unsigned)((ga >> 32) & 0x01FFFFFFu) | 0x80000000u; // addr[56:32] | type=2
  v8i g1;
  g1[0] = (int)(1u << 16);                               // data_size=1 -> 2 bytes
  g1[1] = (int)((tensor_d0 & 0xFFFFu) << 16);            // tensor_dim0[15:0]
  g1[2] = (int)(((tensor_d0 >> 16) & 0xFFFFu) | ((tensor_d1 & 0xFFFFu) << 16));
  g1[3] = (int)(((tensor_d1 >> 16) & 0xFFFFu) | ((tile_d0 & 0xFFFFu) << 16));
  g1[4] = (int)(tile_d1 & 0xFFFFu);                      // tile_dim1 (tile_dim2=0)
  g1[5] = (int)(unsigned)(stride0 & 0xFFFFFFFFull);      // tensor_dim0_stride lo32
  g1[6] = (int)(unsigned)((stride0 >> 32) & 0xFFFFull);  // stride hi16 (dim1_stride=0)
  g1[7] = 0;
  v4i z4 = {0, 0, 0, 0};
  v8i z8 = {0, 0, 0, 0, 0, 0, 0, 0};
  __builtin_amdgcn_tensor_load_to_lds(g0, g1, z4, z4, z8, 0);
}
#endif

// ---------------- conversion kernels ----------------
__global__ void cvt_bf16(const float* __restrict__ src,
                         unsigned short* __restrict__ dst, int n) {
  int i = blockIdx.x * blockDim.x + threadIdx.x;
  int stride = gridDim.x * blockDim.x;
  for (; i < n; i += stride) dst[i] = f2bf(src[i]);
}

__global__ void cvt_qkv_w(const float* __restrict__ wq, const float* __restrict__ wk,
                          const float* __restrict__ wv, unsigned short* __restrict__ dst) {
  int i = blockIdx.x * blockDim.x + threadIdx.x;
  int stride = gridDim.x * blockDim.x;
  const int n = DMODEL * 3 * DMODEL;
  for (; i < n; i += stride) {
    int kk = i / (3 * DMODEL);
    int c  = i % (3 * DMODEL);
    int proj = c >> 10, cc = c & 1023;
    const float* w = (proj == 0) ? wq : ((proj == 1) ? wk : wv);
    dst[i] = f2bf(w[kk * DMODEL + cc]);
  }
}

// ---------------- GEMM: C = A(bf16,[M][K]) * B(bf16,[K][N]) + bias ----------------
// Double-buffered: A via per-lane async-to-LDS, B via TDM (fallback async).
// MODE 0: scatter bf16 into Q/K/V [B,H,S,DK];  MODE 1: f32 output + bias.
template <int MODE>
__global__ __launch_bounds__(256) void gemm_bf16(
    const unsigned short* __restrict__ A, const unsigned short* __restrict__ Bm,
    int M, int N, int K,
    unsigned short* __restrict__ qo, unsigned short* __restrict__ ko,
    unsigned short* __restrict__ vo,
    const float* __restrict__ bq, const float* __restrict__ bk,
    const float* __restrict__ bv,
    float* __restrict__ outf, const float* __restrict__ bo) {
  extern __shared__ char smem[];
  unsigned short* Alds = (unsigned short*)smem;   // 2 bufs x 128 x 72 halves
  unsigned short* Blds = Alds + 2 * 128 * 72;     // 2 bufs x 64 x 64 halves (row-major)
  const unsigned ABytes = 128 * 72 * 2;           // per buffer
  const unsigned BBase  = 2 * ABytes;
  const unsigned BBytes = 64 * 64 * 2;

  const int t = threadIdx.x;
  const int lane = t & 31, wid = t >> 5;
  const int rb = blockIdx.x * 128;
  const int cb = blockIdx.y * 64;
  const int m0 = (wid >> 1) * 32;
  const int n0 = (wid & 1) * 32;

  v8f acc[2][2];
#pragma unroll
  for (int im = 0; im < 2; ++im)
#pragma unroll
    for (int in = 0; in < 2; ++in)
#pragma unroll
      for (int e = 0; e < 8; ++e) acc[im][in][e] = 0.0f;

  auto stage = [&](int buf, int kt) {
    unsigned abase = (unsigned)buf * ABytes;
#pragma unroll
    for (int i = 0; i < 4; ++i) {            // A tile 128x64: 1024 x 16B chunks
      int e = t + i * 256;
      int row = e >> 3, c = e & 7;
      async_b128(abase + row * 144 + c * 16,
                 A + (size_t)(rb + row) * K + kt + c * 8);
    }
#if HAS_TDM
    if (wid == 0)
      tdm_load_2d(BBase + (unsigned)buf * BBytes, Bm + (size_t)kt * N + cb,
                  (unsigned)N, (unsigned)K, 64, 64, (unsigned long long)N);
#else
#pragma unroll
    for (int i = 0; i < 2; ++i) {            // B tile 64x64: 512 x 16B chunks
      int e = t + i * 256;
      int row = e >> 3, c = e & 7;
      async_b128(BBase + (unsigned)buf * BBytes + row * 128 + c * 16,
                 Bm + (size_t)(kt + row) * N + cb + c * 8);
    }
#endif
  };
  auto waitStage = [&]() {
    wait_async0();
#if HAS_TDM
    if (wid == 0) __builtin_amdgcn_s_wait_tensorcnt(0);
#endif
    __syncthreads();
  };

  const int arow = lane & 15, kb = (lane >> 4) * 8;
  const int bcol = lane & 15, kb2 = (lane >> 4) * 16;

  stage(0, 0);
  waitStage();
  for (int kt = 0; kt < K; kt += 64) {
    int cur = (kt >> 6) & 1;
    if (kt + 64 < K) stage(cur ^ 1, kt + 64);

    const unsigned short* Ab = Alds + cur * 128 * 72;
    const unsigned short* Bb = Blds + cur * 64 * 64;
#pragma unroll
    for (int ks = 0; ks < 64; ks += 32) {
      Frag16 af[2], bf2[2];
#pragma unroll
      for (int im = 0; im < 2; ++im) {
        const unsigned short* base = Ab + (m0 + im * 16 + arow) * 72 + ks;
#pragma unroll
        for (int v = 0; v < 8; ++v) {
          int kk = ((v < 4) ? 0 : 16) + kb + (v & 3) * 2;
          af[im].u[v] = *(const unsigned int*)(base + kk);
        }
      }
#pragma unroll
      for (int in = 0; in < 2; ++in) {
        int ncol = n0 + in * 16 + bcol;
#pragma unroll
        for (int v = 0; v < 8; ++v) {      // read-time transpose gather
          int k0 = ks + ((v < 4) ? 0 : 16) + kb + (v & 3) * 2;
          unsigned p0 = Bb[k0 * 64 + ncol];
          unsigned p1 = Bb[(k0 + 1) * 64 + ncol];
          bf2[in].u[v] = p0 | (p1 << 16);
        }
      }
#pragma unroll
      for (int im = 0; im < 2; ++im)
#pragma unroll
        for (int in = 0; in < 2; ++in)
          acc[im][in] = wmma_bf16(af[im], bf2[in], acc[im][in]);
    }
    waitStage();
  }

  const int half = lane >> 4;
#pragma unroll
  for (int im = 0; im < 2; ++im) {
#pragma unroll
    for (int in = 0; in < 2; ++in) {
#pragma unroll
      for (int r = 0; r < 8; ++r) {
        int gr = rb + m0 + im * 16 + r + half * 8;
        int gc = cb + n0 + in * 16 + (lane & 15);
        float val = acc[im][in][r];
        if (MODE == 0) {
          int proj = gc >> 10, d = gc & 1023, h = d >> 6, dk = d & 63;
          const float* bias = (proj == 0) ? bq : ((proj == 1) ? bk : bv);
          unsigned short* dst = (proj == 0) ? qo : ((proj == 1) ? ko : vo);
          val += bias[d];
          int bidx = gr >> 11, s = gr & 2047;
          dst[(((size_t)(bidx * NHEADS + h)) * SEQ + s) * DK + dk] = f2bf(val);
        } else {
          outf[(size_t)gr * N + gc] = val + bo[gc];
        }
      }
    }
  }
}

// ---------------- flash attention: one block = (b,h) x 64 q rows ----------------
__global__ __launch_bounds__(128) void attn_kernel(
    const unsigned short* __restrict__ Qb, const unsigned short* __restrict__ Kb,
    const unsigned short* __restrict__ Vb, const int* __restrict__ mask,
    unsigned short* __restrict__ Ctx) {
  extern __shared__ char smem[];
  unsigned short* Klds = (unsigned short*)smem;   // 2 bufs x 32 x 64 halves
  unsigned short* Vlds = Klds + 2 * 32 * 64;      // 2 bufs x 32 x 64 halves
  unsigned short* Plds = Vlds + 2 * 32 * 64;      // 4 waves x 16 x 40
  const unsigned KVBytes = 32 * 64 * 2;           // per buffer
  const unsigned VBase   = 2 * KVBytes;

  const int t = threadIdx.x, lane = t & 31, wid = t >> 5;
  const int bh = blockIdx.x;
  const int b  = bh >> 4;
  const int qb0 = blockIdx.y * 64 + wid * 16;
  const unsigned short* Qh = Qb + (size_t)bh * SEQ * DK;
  const unsigned short* Kh = Kb + (size_t)bh * SEQ * DK;
  const unsigned short* Vh = Vb + (size_t)bh * SEQ * DK;

  // preload Q A-fragments (features 0..31 / 32..63)
  Frag16 qa0, qa1;
  {
    int qrow = qb0 + (lane & 15);
    const unsigned short* src = Qh + (size_t)qrow * DK;
    int kb = (lane >> 4) * 8;
#pragma unroll
    for (int v = 0; v < 8; ++v) {
      int kk = 2 * kb + v * 2;
      qa0.u[v] = *(const unsigned int*)(src + kk);
      qa1.u[v] = *(const unsigned int*)(src + kk + 32);
    }
  }

  v8f O[4];
  float mrow[8], lrow[8];
#pragma unroll
  for (int nt = 0; nt < 4; ++nt)
#pragma unroll
    for (int e = 0; e < 8; ++e) O[nt][e] = 0.0f;
#pragma unroll
  for (int r = 0; r < 8; ++r) { mrow[r] = -1e30f; lrow[r] = 0.0f; }

  auto stageKV = [&](int buf, int kc) {
#if HAS_TDM
    if (wid == 0) {
      tdm_load_2d((unsigned)buf * KVBytes, Kh + (size_t)kc * DK,
                  DK, SEQ, DK, 32, DK);
      tdm_load_2d(VBase + (unsigned)buf * KVBytes, Vh + (size_t)kc * DK,
                  DK, SEQ, DK, 32, DK);
    }
#else
#pragma unroll
    for (int i = 0; i < 2; ++i) {           // 32x64 halves: 256 x 16B chunks each
      int e = t + i * 128;
      int row = e >> 3, c = e & 7;
      async_b128((unsigned)buf * KVBytes + row * 128 + c * 16,
                 Kh + (size_t)(kc + row) * DK + c * 8);
      async_b128(VBase + (unsigned)buf * KVBytes + row * 128 + c * 16,
                 Vh + (size_t)(kc + row) * DK + c * 8);
    }
#endif
  };
  auto waitKV = [&]() {
#if HAS_TDM
    if (wid == 0) __builtin_amdgcn_s_wait_tensorcnt(0);
#else
    wait_async0();
#endif
    __syncthreads();
  };

  const int kb2 = (lane >> 4) * 16;

  stageKV(0, 0);
  waitKV();
  for (int kc = 0; kc < SEQ; kc += 32) {
    int cur = (kc >> 5) & 1;
    if (kc + 32 < SEQ) stageKV(cur ^ 1, kc + 32);

    const unsigned short* Kc = Klds + cur * 32 * 64;
    const unsigned short* Vc = Vlds + cur * 32 * 64;

    // scores S = Q @ K^T for 32 keys
    v8f s[2];
#pragma unroll
    for (int nt = 0; nt < 2; ++nt) {
      Frag16 kf0, kf1;
      const unsigned short* base = Kc + (nt * 16 + (lane & 15)) * 64 + kb2;
#pragma unroll
      for (int v = 0; v < 8; ++v) {
        kf0.u[v] = *(const unsigned int*)(base + v * 2);
        kf1.u[v] = *(const unsigned int*)(base + 32 + v * 2);
      }
      v8f z;
#pragma unroll
      for (int e = 0; e < 8; ++e) z[e] = 0.0f;
      z = wmma_bf16(qa0, kf0, z);
      z = wmma_bf16(qa1, kf1, z);
      s[nt] = z;
    }

    // scale + mask
    int key0 = kc + (lane & 15);
    bool mk0 = (mask[b * SEQ + key0] == 0);
    bool mk1 = (mask[b * SEQ + key0 + 16] == 0);
#pragma unroll
    for (int r = 0; r < 8; ++r) {
      s[0][r] = mk0 ? -1e9f : s[0][r] * 0.125f;
      s[1][r] = mk1 ? -1e9f : s[1][r] * 0.125f;
    }

    // online softmax
#pragma unroll
    for (int r = 0; r < 8; ++r) {
      float cm = fmaxf(s[0][r], s[1][r]);
#pragma unroll
      for (int off = 1; off < 16; off <<= 1)
        cm = fmaxf(cm, __shfl_xor(cm, off, 32));
      float mn = fmaxf(mrow[r], cm);
      float alpha = __expf(mrow[r] - mn);
      mrow[r] = mn;
      float p0 = __expf(s[0][r] - mn);
      float p1 = __expf(s[1][r] - mn);
      s[0][r] = p0; s[1][r] = p1;
      float rs = p0 + p1;
#pragma unroll
      for (int off = 1; off < 16; off <<= 1)
        rs += __shfl_xor(rs, off, 32);
      lrow[r] = lrow[r] * alpha + rs;
#pragma unroll
      for (int nt = 0; nt < 4; ++nt) O[nt][r] *= alpha;
    }

    // P (C-frag f32) -> A-frag bf16 via per-wave LDS round trip
    unsigned short* Pw = Plds + wid * 16 * 40;
    {
      const int half = lane >> 4;
#pragma unroll
      for (int r = 0; r < 8; ++r) {
        int row = r + half * 8;
        Pw[row * 40 + (lane & 15)]      = f2bf(s[0][r]);
        Pw[row * 40 + 16 + (lane & 15)] = f2bf(s[1][r]);
      }
    }
    wait_ds0();  // same-wave cross-lane LDS hazard

    Frag16 pf;
    {
      int prow = lane & 15, kb = (lane >> 4) * 8;
      const unsigned short* base = Pw + prow * 40;
#pragma unroll
      for (int v = 0; v < 8; ++v) {
        int kk = 2 * kb + v * 2;
        pf.u[v] = *(const unsigned int*)(base + kk);
      }
    }

    // O += P @ V (read-time transpose gather of V)
#pragma unroll
    for (int nt = 0; nt < 4; ++nt) {
      Frag16 vf;
      int dim = nt * 16 + (lane & 15);
#pragma unroll
      for (int v = 0; v < 8; ++v) {
        int k0 = kb2 + 2 * v;
        unsigned p0 = Vc[k0 * 64 + dim];
        unsigned p1 = Vc[(k0 + 1) * 64 + dim];
        vf.u[v] = p0 | (p1 << 16);
      }
      O[nt] = wmma_bf16(pf, vf, O[nt]);
    }
    waitKV();
  }

  const int half = lane >> 4;
#pragma unroll
  for (int r = 0; r < 8; ++r) {
    float inv = 1.0f / fmaxf(lrow[r], 1e-20f);
    int qrow = qb0 + r + half * 8;
    size_t base = ((size_t)b * SEQ + qrow) * DMODEL + (bh & 15) * DK + (lane & 15);
#pragma unroll
    for (int nt = 0; nt < 4; ++nt)
      Ctx[base + nt * 16] = f2bf(O[nt][r] * inv);
  }
}

// ---------------- launch ----------------
extern "C" void kernel_launch(void* const* d_in, const int* in_sizes, int n_in,
                              void* d_out, int out_size, void* d_ws, size_t ws_size,
                              hipStream_t stream) {
  const float* x  = (const float*)d_in[0];
  const int* mask = (const int*)d_in[1];
  const float* wq = (const float*)d_in[2];
  const float* bq = (const float*)d_in[3];
  const float* wk = (const float*)d_in[4];
  const float* bk = (const float*)d_in[5];
  const float* wv = (const float*)d_in[6];
  const float* bv = (const float*)d_in[7];
  const float* wo = (const float*)d_in[8];
  const float* bo = (const float*)d_in[9];
  float* out = (float*)d_out;

  unsigned short* Xb   = (unsigned short*)d_ws;                 // [8192][1024]
  unsigned short* Wqkv = Xb   + (size_t)MTOT * DMODEL;          // [1024][3072]
  unsigned short* Wo   = Wqkv + (size_t)DMODEL * 3 * DMODEL;    // [1024][1024]
  unsigned short* Qb   = Wo   + (size_t)DMODEL * DMODEL;        // [B,H,S,DK]
  unsigned short* Kb   = Qb   + (size_t)MTOT * DMODEL;
  unsigned short* Vb   = Kb   + (size_t)MTOT * DMODEL;
  unsigned short* Ctx  = Vb   + (size_t)MTOT * DMODEL;          // [8192][1024]

  cvt_bf16<<<2048, 256, 0, stream>>>(x, Xb, MTOT * DMODEL);
  cvt_qkv_w<<<2048, 256, 0, stream>>>(wq, wk, wv, Wqkv);
  cvt_bf16<<<1024, 256, 0, stream>>>(wo, Wo, DMODEL * DMODEL);

  const size_t gemm_smem = (size_t)(2 * 128 * 72 + 2 * 64 * 64) * 2;  // 53248 B
  dim3 g1(MTOT / 128, (3 * DMODEL) / 64);  // 64 x 48
  gemm_bf16<0><<<g1, 256, gemm_smem, stream>>>(Xb, Wqkv, MTOT, 3 * DMODEL, DMODEL,
                                               Qb, Kb, Vb, bq, bk, bv, nullptr, nullptr);

  dim3 g2(BATCH * NHEADS, SEQ / 64);       // 64 x 32
  const size_t attn_smem = (size_t)(2 * 32 * 64 * 2 + 4 * 16 * 40) * 2; // 21504 B
  attn_kernel<<<g2, 128, attn_smem, stream>>>(Qb, Kb, Vb, mask, Ctx);

  dim3 g3(MTOT / 128, DMODEL / 64);        // 64 x 16
  gemm_bf16<1><<<g3, 256, gemm_smem, stream>>>(Ctx, Wo, MTOT, DMODEL, DMODEL,
                                               nullptr, nullptr, nullptr,
                                               nullptr, nullptr, nullptr, out, bo);
}
